// SelfAttention_49916109914344
// MI455X (gfx1250) — matrix operand order repaired and checked by reading, not compiled, not run on hardware
//
#include <hip/hip_runtime.h>
#include <hip/hip_bf16.h>

// ---------------------------------------------------------------------------
// Self-attention (softmax over dim=1) fused flash-style for MI455X (gfx1250).
// B=8, C=512, N=64*64=4096, D=64.  Matmuls on v_wmma_f32_16x16x32_f16;
// V-chunk streaming via Tensor Data Mover (tensor_load_to_lds), double-buffered.
// ---------------------------------------------------------------------------

typedef __attribute__((ext_vector_type(16))) _Float16 v16h;
typedef __attribute__((ext_vector_type(8)))  _Float16 v8h;
typedef __attribute__((ext_vector_type(8)))  float    v8f;
typedef unsigned int u32;
typedef __attribute__((ext_vector_type(4)))  u32      v4u;
typedef __attribute__((ext_vector_type(8)))  int      v8i;
typedef __attribute__((ext_vector_type(4)))  int      v4i;

#define B_    8
#define C_    512
#define N_    4096
#define D_    64
#define MPROJ 640   // stacked rows: q(0..63), k(64..127), v(128..639)

#if __has_builtin(__builtin_amdgcn_tensor_load_to_lds) && \
    __has_builtin(__builtin_amdgcn_s_wait_tensorcnt)
#define HAVE_TDM 1
#else
#define HAVE_TDM 0
#endif

__device__ __forceinline__ v8f wmma16(v16h a, v16h b, v8f c) {
  // D = A(16x32 f16) * B(32x16 f16) + C(16x16 f32)
  return __builtin_amdgcn_wmma_f32_16x16x32_f16(
      /*neg_a=*/false, a, /*neg_b=*/false, b,
      /*c_mod=*/(short)0, c, /*reuse_a=*/false, /*reuse_b=*/false);
}

// 16-bit A-matrix 16x32 layout: lanes 0-15 hold K{0..7,16..23}, lanes 16-31
// hold K{8..15,24..31}; row M = lane&15.
__device__ __forceinline__ v16h a_frag(const _Float16* row, int lane) {
  const int kb = (lane >> 4) * 8;
  union { v16h v; v8h h[2]; } u;
  u.h[0] = *(const v8h*)(row + kb);        // VGPR0..3 : K = kb..kb+7
  u.h[1] = *(const v8h*)(row + kb + 16);   // VGPR4..7 : K = kb+16..kb+23
  return u.v;
}

// 16-bit B-matrix 32x16 layout: col N = lane&15; lanes 0-15 hold K{0..7,16..23},
// lanes 16-31 hold K{8..15,24..31} (same K order as a_frag).
__device__ __forceinline__ v16h b_frag(const _Float16* row, int lane) {
  const int kb = (lane >> 4) * 8;
  union { v16h v; v8h h[2]; } u;
  u.h[0] = *(const v8h*)(row + kb);
  u.h[1] = *(const v8h*)(row + kb + 16);
  return u.v;
}

#if HAVE_TDM
// Issue a 2D TDM tile load: tile (tile_d1 rows x tile_d0 elems) of 16-bit
// data from a row-major tensor (row stride = stride0 elems) into LDS at
// byte offset lds_off.  D# layout per CDNA5 ISA ch.8 (groups 0/1; 2D).
__device__ __forceinline__ void tdm_load_2d_f16(
    unsigned lds_off, const void* gptr,
    u32 tensor_d0, u32 tensor_d1, u32 tile_d0, u32 tile_d1, u32 stride0) {
  const unsigned long long ga = (unsigned long long)(uintptr_t)gptr;
  const v4u g0 = { 1u,                                   // count=1, user D#
                   lds_off,                              // lds_addr (bytes)
                   (u32)ga,                              // global_addr[31:0]
                   (u32)((ga >> 32) & 0x01ffffffu) | (2u << 30) }; // [56:32]|type=2
  const v8i g1 = { (int)(1u << 16),                      // data_size=1 -> 2B
                   (int)((tensor_d0 & 0xffffu) << 16),   // tensor_dim0[15:0]
                   (int)((tensor_d0 >> 16) | ((tensor_d1 & 0xffffu) << 16)),
                   (int)((tensor_d1 >> 16) | (tile_d0 << 16)),
                   (int)tile_d1,                         // tile_dim1; tile_dim2=0
                   (int)stride0,                         // dim0_stride[31:0]
                   0, 0 };
  const v4i z4 = {0, 0, 0, 0};
#if defined(__clang_major__) && (__clang_major__ >= 23)
  const v8i z8 = {0, 0, 0, 0, 0, 0, 0, 0};
  __builtin_amdgcn_tensor_load_to_lds(g0, g1, z4, z4, z8, 0);
#else
  __builtin_amdgcn_tensor_load_to_lds(g0, g1, z4, z4, 0);
#endif
}
#endif

// ---------------------------------------------------------------------------
// Projection GEMM: Y[b][m][n] = Wcat[m][:] . x[b][:][n],  m<640, f16 output.
// WG = 256 threads (8 waves, 4x2), tile 64(M) x 128(N), K-steps of 32.
// ---------------------------------------------------------------------------
struct alignas(128) ProjLDS {
  _Float16 lA[64][32];    // Wcat tile, rows m, contiguous k
  _Float16 lB[128][32];   // x tile transposed: [n][k]
};

__global__ __launch_bounds__(256)
void proj_kernel(const float* __restrict__ x,  const float* __restrict__ Wq,
                 const float* __restrict__ Wk, const float* __restrict__ Wv,
                 _Float16* __restrict__ yh) {
  __shared__ ProjLDS s;
  const int tid = threadIdx.x, lane = tid & 31, wv = tid >> 5;
  const int wm = wv >> 1, wn = wv & 1;
  const int nBase = blockIdx.x * 128;
  const int mBase = blockIdx.y * 64;
  const int b     = blockIdx.z;

  const v8f z = {0.f,0.f,0.f,0.f,0.f,0.f,0.f,0.f};
  v8f acc[4];
  #pragma unroll
  for (int t = 0; t < 4; ++t) acc[t] = z;

  const float* xb = x + (size_t)b * C_ * N_;

  for (int kb = 0; kb < C_; kb += 32) {
    { // A tile: 64 rows x 32 k, selecting Wq/Wk/Wv by stacked row, f32->f16
      const int r  = tid >> 2;
      const int k0 = (tid & 3) * 8;
      const int m  = mBase + r;
      const float* wr = (m < 64)  ? (Wq + (size_t)m * C_)
                      : (m < 128) ? (Wk + (size_t)(m - 64) * C_)
                                  : (Wv + (size_t)(m - 128) * C_);
      #pragma unroll
      for (int i = 0; i < 8; ++i)
        s.lA[r][k0 + i] = (_Float16)wr[kb + k0 + i];
    }
    { // B tile transposed: lB[n][k] = x[b][kb+k][nBase+n]  (coalesced over n)
      const int n  = tid & 127;
      const int k0 = (tid >> 7) * 16;
      #pragma unroll
      for (int i = 0; i < 16; ++i)
        s.lB[n][k0 + i] = (_Float16)xb[(size_t)(kb + k0 + i) * N_ + nBase + n];
    }
    __syncthreads();

    const v16h aF = a_frag(&s.lA[wm * 16 + (lane & 15)][0], lane);
    #pragma unroll
    for (int t = 0; t < 4; ++t) {
      const v16h bF = b_frag(&s.lB[wn * 64 + t * 16 + (lane & 15)][0], lane);
      acc[t] = wmma16(aF, bF, acc[t]);
    }
    __syncthreads();
  }

  _Float16* yb = yh + (size_t)b * MPROJ * N_;
  #pragma unroll
  for (int t = 0; t < 4; ++t) {
    const int ncol = nBase + wn * 64 + t * 16 + (lane & 15);
    #pragma unroll
    for (int j = 0; j < 8; ++j) {
      const int mrow = mBase + wm * 16 + j + ((lane >> 4) ? 8 : 0);
      yb[(size_t)mrow * N_ + ncol] = (_Float16)acc[t][j];
    }
  }
}

// ---------------------------------------------------------------------------
// Fused flash attention:
//   S[n,m] = sum_d q[d,n] k[d,m] ; P = softmax over n (online) ;
//   O[c,m] = sum_n v[c,n] P[n,m] ; out = gamma*O + x.
// WG = 256 threads; owns (b, 64 cols m, 256 channels c); streams n by 32.
// V chunks arrive via TDM (double-buffered); q/k transposes staged manually.
// ---------------------------------------------------------------------------
struct alignas(128) FlashLDS {
  _Float16 lV[2][256][32];  // [buf][c][n]  (TDM destination, double-buffered)
  _Float16 kT[64][64];      // [m][d]
  _Float16 qT[32][64];      // [n][d]   (transpose of q chunk)
  _Float16 pT[64][32];      // [m][n]   (transpose of P chunk)
  float colmax2[2][64];
  float colsum2[2][64];
  float runmax[64];
  float runsum[64];
  float scaleA[64];
};

__global__ __launch_bounds__(256)
void flash_kernel(const _Float16* __restrict__ yh, const float* __restrict__ x,
                  const float* __restrict__ gamma, float* __restrict__ out) {
  __shared__ FlashLDS s;
  const int tid  = threadIdx.x, lane = tid & 31, wv = tid >> 5;
  const int msub = wv & 3;
  const int half = wv >> 2;     // nsub for S, chalf for O
  const int cBase = blockIdx.x * 256;
  const int mBase = blockIdx.y * 64;
  const int b     = blockIdx.z;

  const _Float16* qh = yh + (size_t)b * MPROJ * N_;
  const _Float16* kh = qh + (size_t)D_ * N_;
  const _Float16* vh = qh + (size_t)(2 * D_) * N_;
  const _Float16* vtile = vh + (size_t)cBase * N_;

#if HAVE_TDM
  // LDS byte offsets of the two V buffers (struct assumed at LDS offset 0).
  const unsigned lvOff0 = (unsigned)((char*)&s.lV[0][0][0] - (char*)&s);
  const unsigned lvOff1 = (unsigned)((char*)&s.lV[1][0][0] - (char*)&s);
  if (wv == 0)   // chunk 0 -> buffer 0 (issued once per wave; wave 0 only)
    tdm_load_2d_f16(lvOff0, vtile, N_, C_, 32, 256, N_);
#endif

  if (tid < 64) { s.runmax[tid] = -3.0e38f; s.runsum[tid] = 0.0f; }
  #pragma unroll
  for (int i = 0; i < 16; ++i) {          // K tile: kT[m][d] = k[d][mBase+m]
    const int idx = tid + i * 256;
    const int d = idx >> 6, m = idx & 63;
    s.kT[m][d] = kh[(size_t)d * N_ + mBase + m];
  }

  const v8f z = {0.f,0.f,0.f,0.f,0.f,0.f,0.f,0.f};
  v8f accO[8];
  #pragma unroll
  for (int ct = 0; ct < 8; ++ct) accO[ct] = z;

  __syncthreads();

  for (int nb = 0; nb < N_; nb += 32) {
    const int buf = (nb >> 5) & 1;

#if HAVE_TDM
    if (wv == 0 && nb + 32 < N_)          // async-prefetch next V chunk
      tdm_load_2d_f16(buf ? lvOff0 : lvOff1, vtile + (nb + 32),
                      N_, C_, 32, 256, N_);
#endif
    if (nb + 32 < N_)                     // -> global_prefetch_b8
      __builtin_prefetch(qh + (size_t)lane * N_ + nb + 32, 0, 1);

    #pragma unroll
    for (int i = 0; i < 8; ++i) {         // qT[n][d] = q[d][nb+n]
      const int idx = tid + i * 256;
      const int d = idx >> 5, n = idx & 31;
      s.qT[n][d] = qh[(size_t)d * N_ + nb + n];
    }
#if HAVE_TDM
    if (wv == 0) {                        // current buffer's TDM is the older op
      if (nb + 32 < N_) __builtin_amdgcn_s_wait_tensorcnt((short)1);
      else              __builtin_amdgcn_s_wait_tensorcnt((short)0);
    }
#else
    #pragma unroll
    for (int i = 0; i < 32; ++i) {        // fallback: manual V chunk load
      const int idx = tid + i * 256;
      const int c = idx >> 5, n = idx & 31;
      s.lV[buf][c][n] = vtile[(size_t)c * N_ + nb + n];
    }
#endif
    __syncthreads();

    // ---- S chunk: one 16x16 tile per wave, K = 64 over d ----
    v8f accS = z;
    #pragma unroll
    for (int dwin = 0; dwin < 64; dwin += 32) {
      const v16h aS = a_frag(&s.qT[half * 16 + (lane & 15)][dwin], lane);
      const v16h bS = b_frag(&s.kT[msub * 16 + (lane & 15)][dwin], lane);
      accS = wmma16(aS, bS, accS);
    }

    // ---- column max over this wave's 16 rows (regs + xor-16 shuffle) ----
    float pm = accS[0];
    #pragma unroll
    for (int j = 1; j < 8; ++j) pm = fmaxf(pm, accS[j]);
    pm = fmaxf(pm, __shfl_xor(pm, 16, 32));
    if (lane < 16) s.colmax2[half][msub * 16 + lane] = pm;
    __syncthreads();

    if (tid < 64) {                       // online-max update, per column m
      const float cm = fmaxf(s.colmax2[0][tid], s.colmax2[1][tid]);
      const float om = s.runmax[tid];
      const float nm = fmaxf(om, cm);
      s.scaleA[tid] = __expf(om - nm);
      s.runmax[tid] = nm;
    }
    __syncthreads();

    { // ---- P = exp(S - newmax), store transposed, column partial sums ----
      const float nm = s.runmax[msub * 16 + (lane & 15)];
      float ps = 0.0f;
      #pragma unroll
      for (int j = 0; j < 8; ++j) {
        const float p = __expf(accS[j] - nm);
        ps += p;
        s.pT[msub * 16 + (lane & 15)]
            [half * 16 + j + ((lane >> 4) ? 8 : 0)] = (_Float16)p;
      }
      ps += __shfl_xor(ps, 16, 32);
      if (lane < 16) s.colsum2[half][msub * 16 + lane] = ps;
    }
    __syncthreads();

    if (tid < 64)
      s.runsum[tid] = s.runsum[tid] * s.scaleA[tid]
                    + s.colsum2[0][tid] + s.colsum2[1][tid];

    { // ---- rescale O, then O += V * P (8 WMMAs per wave, K = 32) ----
      const float sc = s.scaleA[msub * 16 + (lane & 15)];
      #pragma unroll
      for (int ct = 0; ct < 8; ++ct)
        #pragma unroll
        for (int j = 0; j < 8; ++j) accO[ct][j] *= sc;

      const v16h bP = b_frag(&s.pT[msub * 16 + (lane & 15)][0], lane);
      #pragma unroll
      for (int ct = 0; ct < 8; ++ct) {
        const v16h aV =
            a_frag(&s.lV[buf][half * 128 + ct * 16 + (lane & 15)][0], lane);
        accO[ct] = wmma16(aV, bP, accO[ct]);
      }
    }
    __syncthreads();
  }

  // ---- epilogue: out = gamma * O / l + x ----
  const float g   = gamma[0];
  const float inv = 1.0f / s.runsum[msub * 16 + (lane & 15)];
  const int mcol  = mBase + msub * 16 + (lane & 15);
  const float* xb = x   + (size_t)b * C_ * N_;
  float*       ob = out + (size_t)b * C_ * N_;
  #pragma unroll
  for (int ct = 0; ct < 8; ++ct) {
    #pragma unroll
    for (int j = 0; j < 8; ++j) {
      const int c = cBase + half * 128 + ct * 16 + j + ((lane >> 4) ? 8 : 0);
      const size_t off = (size_t)c * N_ + mcol;
      ob[off] = g * accO[ct][j] * inv + xb[off];
    }
  }
}

// ---------------------------------------------------------------------------
extern "C" void kernel_launch(void* const* d_in, const int* in_sizes, int n_in,
                              void* d_out, int out_size, void* d_ws, size_t ws_size,
                              hipStream_t stream) {
  (void)in_sizes; (void)n_in; (void)out_size; (void)ws_size;
  const float* x     = (const float*)d_in[0];
  const float* Wq    = (const float*)d_in[1];
  const float* Wk    = (const float*)d_in[2];
  const float* Wv    = (const float*)d_in[3];
  const float* gamma = (const float*)d_in[4];
  float* out = (float*)d_out;

  // scratch: stacked f16 projections y[b][640][4096]  (~40 MB)
  _Float16* yh = (_Float16*)d_ws;

  proj_kernel<<<dim3(N_ / 128, MPROJ / 64, B_), 256, 0, stream>>>(
      x, Wq, Wk, Wv, yh);
  flash_kernel<<<dim3(C_ / 256, N_ / 64, B_), 256, 0, stream>>>(
      yh, x, gamma, out);
}
